// GCNNetwork_23356032156067
// MI455X (gfx1250) — matrix operand order repaired and checked by reading, not compiled, not run on hardware
//
#include <hip/hip_runtime.h>

#define NN  8192
#define FIN 512
#define HID 256
#define CLS 16
#define AP  40   // padded LDS row length in u16 (80B: conflict-free b128 row reads)

typedef unsigned int   u32;
typedef unsigned short u16;
typedef __bf16 bf16;
typedef __attribute__((ext_vector_type(4)))  u32   u32x4;
typedef __attribute__((ext_vector_type(8)))  bf16  v8bf;
typedef __attribute__((ext_vector_type(16))) bf16  v16bf;
typedef __attribute__((ext_vector_type(8)))  float v8f;
typedef __attribute__((ext_vector_type(4)))  float f32x4;

#if defined(__gfx1250__) &&                                            \
    __has_builtin(__builtin_amdgcn_global_load_async_to_lds_b128) &&   \
    __has_builtin(__builtin_amdgcn_s_wait_asynccnt)
#define USE_ASYNC 1
#define AS1 __attribute__((address_space(1)))
#define AS3 __attribute__((address_space(3)))
typedef int i4v __attribute__((vector_size(16)));   // matches builtin's v4i param
#define GLD_ASYNC_B128(gsrc, ldst) \
  __builtin_amdgcn_global_load_async_to_lds_b128((AS1 i4v*)(gsrc), (AS3 i4v*)(ldst), 0, 0)
#else
#define USE_ASYNC 0
#endif

// Native bf16 converts (backend emits v_cvt_*bf16_f32 instead of bfe/add3 chains)
static __device__ __forceinline__ u16 f2bf_bits(float f) {
  return __builtin_bit_cast(u16, static_cast<bf16>(f));
}
static __device__ __forceinline__ u32 pk2(float lo, float hi) {
  return (u32)f2bf_bits(lo) | ((u32)f2bf_bits(hi) << 16);
}
static __device__ __forceinline__ v8bf ld8(const u16* p) {   // 16B LDS/global read
  return __builtin_bit_cast(v8bf, *(const u32x4*)p);
}
static __device__ __forceinline__ v16bf combine(v8bf lo, v8bf hi) {
  v16bf r;
#pragma unroll
  for (int i = 0; i < 8; ++i) { r[i] = lo[i]; r[i + 8] = hi[i]; }
  return r;
}
static __device__ __forceinline__ v16bf cvt16(f32x4 a, f32x4 b, f32x4 c, f32x4 d) {
  v16bf r;
  r[0]  = (bf16)a.x; r[1]  = (bf16)a.y; r[2]  = (bf16)a.z; r[3]  = (bf16)a.w;
  r[4]  = (bf16)b.x; r[5]  = (bf16)b.y; r[6]  = (bf16)b.z; r[7]  = (bf16)b.w;
  r[8]  = (bf16)c.x; r[9]  = (bf16)c.y; r[10] = (bf16)c.z; r[11] = (bf16)c.w;
  r[12] = (bf16)d.x; r[13] = (bf16)d.y; r[14] = (bf16)d.z; r[15] = (bf16)d.w;
  return r;
}
static __device__ __forceinline__ v8f wmma_bf16(v16bf a, v16bf b, v8f c) {
  return __builtin_amdgcn_wmma_f32_16x16x32_bf16(false, a, false, b, (short)0, c,
                                                 false, false);
}

// ---------------------------------------------------------------------------
// K1: fused degree (d = rsqrt(rowsum(A))) + A f32 -> bf16 conversion, 1 pass.
__global__ __launch_bounds__(256) void k_deg_cvt(const float* __restrict__ A,
                                                 u16* __restrict__ Abf,
                                                 float* __restrict__ dv) {
  const int row = blockIdx.x;
  const float* ar = A + (size_t)row * NN;
  u16* br = Abf + (size_t)row * NN;
  float s = 0.f;
#pragma unroll
  for (int it = 0; it < 4; ++it) {
    int c = (threadIdx.x + it * 256) * 8;
    f32x4 v0 = *(const f32x4*)(ar + c);
    f32x4 v1 = *(const f32x4*)(ar + c + 4);
    s += (v0.x + v0.y) + (v0.z + v0.w) + (v1.x + v1.y) + (v1.z + v1.w);
    u32x4 o;
    o.x = pk2(v0.x, v0.y); o.y = pk2(v0.z, v0.w);
    o.z = pk2(v1.x, v1.y); o.w = pk2(v1.z, v1.w);
    *(u32x4*)(br + c) = o;
  }
#pragma unroll
  for (int o = 16; o > 0; o >>= 1) s += __shfl_xor(s, o, 32);
  __shared__ float red[8];
  if ((threadIdx.x & 31) == 0) red[threadIdx.x >> 5] = s;
  __syncthreads();
  if (threadIdx.x == 0) {
    float t = 0.f;
#pragma unroll
    for (int i = 0; i < 8; ++i) t += red[i];
    dv[row] = rsqrtf(t);
  }
}

// ---------------------------------------------------------------------------
// K2: T1T[n][m] = d[m] * (X @ W1^T)[m][n]   (stored transposed, bf16)
__global__ __launch_bounds__(128) void k_xw1(const float* __restrict__ X,
                                             const float* __restrict__ W1,
                                             const float* __restrict__ dv,
                                             u16* __restrict__ T1T) {
  const int lane = threadIdx.x & 31;
  const int wave = threadIdx.x >> 5;
  const int tile = blockIdx.x * 4 + wave;       // tile = mt*16 + nt
  const int m0 = (tile >> 4) << 4;
  const int n0 = (tile & 15) << 4;
  const int l = lane & 15;
  const int half = lane >> 4;
  v8f acc = {};
  const float* xr = X + (size_t)(m0 + l) * FIN;
  const float* wr = W1 + (size_t)(n0 + l) * FIN;
  for (int k0 = 0; k0 < FIN; k0 += 32) {
    const float* xa = xr + k0 + half * 8;       // A frag: K {0..7,16..23}/{8..15,24..31}
    v16bf af = cvt16(*(const f32x4*)(xa), *(const f32x4*)(xa + 4),
                     *(const f32x4*)(xa + 16), *(const f32x4*)(xa + 20));
    const float* wb = wr + k0 + half * 8;       // B frag: K {0..7,16..23}/{8..15,24..31}
    v16bf bfr = cvt16(*(const f32x4*)(wb), *(const f32x4*)(wb + 4),
                      *(const f32x4*)(wb + 16), *(const f32x4*)(wb + 20));
    acc = wmma_bf16(af, bfr, acc);
  }
  const int n = n0 + l;
#pragma unroll
  for (int v = 0; v < 8; ++v) {
    int m = m0 + v + half * 8;
    T1T[(size_t)n * NN + m] = f2bf_bits(acc[v] * dv[m]);
  }
}

// ---------------------------------------------------------------------------
// K3: H1 = relu(d ⊙ (A_bf16 @ T1)) — main aggregation, BM=64 x BN=256 x BK=32.
// Double-buffered LDS; async-to-LDS DMA when available, reg-prefetch otherwise.
__global__ __launch_bounds__(256) void k_agg1(const u16* __restrict__ Abf,
                                              const u16* __restrict__ T1T,
                                              const float* __restrict__ dv,
                                              u16* __restrict__ H1) {
  __shared__ u16 lA[2][64 * AP];
  __shared__ u16 lB[2][256 * AP];
  const int m0 = blockIdx.x * 64;
  const int lane = threadIdx.x & 31;
  const int wave = threadIdx.x >> 5;
  const int strip = wave >> 1;   // 16-row strip 0..3
  const int nhalf = wave & 1;    // 128-col half
  const int l = lane & 15;
  const int half = lane >> 4;
  v8f acc[8] = {};

  const int arow = threadIdx.x >> 2;
  const int aq = (threadIdx.x & 3) * 8;
  const u16* agp = Abf + (size_t)(m0 + arow) * NN + aq;
  const u16* bgp = T1T + (size_t)threadIdx.x * NN;   // n = threadIdx.x (BN=256)

  const u16* apb = &lA[0][(strip * 16 + l) * AP + half * 8];
  const u16* bpb = &lB[0][(nhalf * 128 + l) * AP + half * 8];
  const int lbuf = 64 * AP;   // lA buffer stride (u16)
  const int bbuf = 256 * AP;  // lB buffer stride (u16)

#if USE_ASYNC
  auto issue = [&](int buf, int k0) {
    GLD_ASYNC_B128(agp + k0, &lA[buf][arow * AP + aq]);
#pragma unroll
    for (int i = 0; i < 4; ++i)
      GLD_ASYNC_B128(bgp + k0 + i * 8, &lB[buf][threadIdx.x * AP + i * 8]);
  };
  issue(0, 0);
  for (int k0 = 0; k0 < NN; k0 += 32) {
    const int cur = (k0 >> 5) & 1;
    __builtin_amdgcn_s_wait_asynccnt(0);   // this wave's DMAs landed
    __syncthreads();                       // everyone's landed; prev compute done
    if (k0 + 32 < NN) issue(cur ^ 1, k0 + 32);
    v16bf af = combine(ld8(apb + cur * lbuf), ld8(apb + cur * lbuf + 16));
#pragma unroll
    for (int nt = 0; nt < 8; ++nt) {
      const u16* bp = bpb + cur * bbuf + nt * 16 * AP;
      acc[nt] = wmma_bf16(af, combine(ld8(bp), ld8(bp + 16)), acc[nt]);
    }
  }
#else
  u32x4 ra, rb0, rb1, rb2, rb3;
  auto fetch = [&](int k0) {
    ra  = *(const u32x4*)(agp + k0);
    rb0 = *(const u32x4*)(bgp + k0);
    rb1 = *(const u32x4*)(bgp + k0 + 8);
    rb2 = *(const u32x4*)(bgp + k0 + 16);
    rb3 = *(const u32x4*)(bgp + k0 + 24);
  };
  fetch(0);
  for (int k0 = 0; k0 < NN; k0 += 32) {
    const int cur = (k0 >> 5) & 1;
    __syncthreads();                       // prev compute done before overwrite
    *(u32x4*)&lA[cur][arow * AP + aq] = ra;
    *(u32x4*)&lB[cur][threadIdx.x * AP]      = rb0;
    *(u32x4*)&lB[cur][threadIdx.x * AP + 8]  = rb1;
    *(u32x4*)&lB[cur][threadIdx.x * AP + 16] = rb2;
    *(u32x4*)&lB[cur][threadIdx.x * AP + 24] = rb3;
    __syncthreads();
    if (k0 + 32 < NN) fetch(k0 + 32);      // overlap next loads with compute
    v16bf af = combine(ld8(apb + cur * lbuf), ld8(apb + cur * lbuf + 16));
#pragma unroll
    for (int nt = 0; nt < 8; ++nt) {
      const u16* bp = bpb + cur * bbuf + nt * 16 * AP;
      acc[nt] = wmma_bf16(af, combine(ld8(bp), ld8(bp + 16)), acc[nt]);
    }
  }
#endif
  float dm[8];
#pragma unroll
  for (int v = 0; v < 8; ++v) dm[v] = dv[m0 + strip * 16 + v + half * 8];
#pragma unroll
  for (int nt = 0; nt < 8; ++nt) {
    int n = nhalf * 128 + nt * 16 + l;
#pragma unroll
    for (int v = 0; v < 8; ++v) {
      int m = m0 + strip * 16 + v + half * 8;
      float val = fmaxf(acc[nt][v] * dm[v], 0.f);   // relu
      H1[(size_t)m * HID + n] = f2bf_bits(val);
    }
  }
}

// ---------------------------------------------------------------------------
// K4: T2T[c][m] = d[m] * (H1 @ W2^T)[m][c]   (transposed, bf16)
__global__ __launch_bounds__(128) void k_hw2(const u16* __restrict__ H1,
                                             const float* __restrict__ W2,
                                             const float* __restrict__ dv,
                                             u16* __restrict__ T2T) {
  const int lane = threadIdx.x & 31;
  const int wave = threadIdx.x >> 5;
  const int m0 = (blockIdx.x * 4 + wave) * 16;
  const int l = lane & 15;
  const int half = lane >> 4;
  v8f acc = {};
  const u16* hr = H1 + (size_t)(m0 + l) * HID;
  const float* wr = W2 + (size_t)l * HID;
  for (int k0 = 0; k0 < HID; k0 += 32) {
    const u16* ha = hr + k0 + half * 8;
    v16bf af = combine(ld8(ha), ld8(ha + 16));
    const float* wb = wr + k0 + half * 8;
    v16bf bfr = cvt16(*(const f32x4*)(wb), *(const f32x4*)(wb + 4),
                      *(const f32x4*)(wb + 16), *(const f32x4*)(wb + 20));
    acc = wmma_bf16(af, bfr, acc);
  }
#pragma unroll
  for (int v = 0; v < 8; ++v) {
    int m = m0 + v + half * 8;
    T2T[(size_t)l * NN + m] = f2bf_bits(acc[v] * dv[m]);
  }
}

// ---------------------------------------------------------------------------
// K5: S = d ⊙ (A_bf16 @ T2) — BM=128, N=16, double-buffered like K3.
__global__ __launch_bounds__(256) void k_agg2(const u16* __restrict__ Abf,
                                              const u16* __restrict__ T2T,
                                              const float* __restrict__ dv,
                                              float* __restrict__ S) {
  __shared__ u16 lA[2][128 * AP];
  __shared__ u16 lB[2][16 * AP];
  const int m0 = blockIdx.x * 128;
  const int lane = threadIdx.x & 31;
  const int wave = threadIdx.x >> 5;
  const int l = lane & 15;
  const int half = lane >> 4;
  v8f acc = {};
  const int arow = threadIdx.x >> 1;
  const int aq = (threadIdx.x & 1) * 16;
  const u16* agp = Abf + (size_t)(m0 + arow) * NN + aq;
  const int bn = threadIdx.x >> 2;
  const int bq = (threadIdx.x & 3) * 8;
  const u16* bgp = T2T + (size_t)bn * NN + bq;

  const u16* apb = &lA[0][(wave * 16 + l) * AP + half * 8];
  const u16* bpb = &lB[0][l * AP + half * 8];
  const int lbuf = 128 * AP;
  const int bbuf = 16 * AP;

#if USE_ASYNC
  auto issue = [&](int buf, int k0) {
    GLD_ASYNC_B128(agp + k0,     &lA[buf][arow * AP + aq]);
    GLD_ASYNC_B128(agp + k0 + 8, &lA[buf][arow * AP + aq + 8]);
    if (threadIdx.x < 64)   // wave-uniform for waves 0/1
      GLD_ASYNC_B128(bgp + k0, &lB[buf][bn * AP + bq]);
  };
  issue(0, 0);
  for (int k0 = 0; k0 < NN; k0 += 32) {
    const int cur = (k0 >> 5) & 1;
    __builtin_amdgcn_s_wait_asynccnt(0);
    __syncthreads();
    if (k0 + 32 < NN) issue(cur ^ 1, k0 + 32);
    v16bf af = combine(ld8(apb + cur * lbuf), ld8(apb + cur * lbuf + 16));
    const u16* bp = bpb + cur * bbuf;
    acc = wmma_bf16(af, combine(ld8(bp), ld8(bp + 16)), acc);
  }
#else
  u32x4 ra0, ra1, rb;
  auto fetch = [&](int k0) {
    ra0 = *(const u32x4*)(agp + k0);
    ra1 = *(const u32x4*)(agp + k0 + 8);
    if (threadIdx.x < 64) rb = *(const u32x4*)(bgp + k0);
  };
  fetch(0);
  for (int k0 = 0; k0 < NN; k0 += 32) {
    const int cur = (k0 >> 5) & 1;
    __syncthreads();
    *(u32x4*)&lA[cur][arow * AP + aq]     = ra0;
    *(u32x4*)&lA[cur][arow * AP + aq + 8] = ra1;
    if (threadIdx.x < 64) *(u32x4*)&lB[cur][bn * AP + bq] = rb;
    __syncthreads();
    if (k0 + 32 < NN) fetch(k0 + 32);
    v16bf af = combine(ld8(apb + cur * lbuf), ld8(apb + cur * lbuf + 16));
    const u16* bp = bpb + cur * bbuf;
    acc = wmma_bf16(af, combine(ld8(bp), ld8(bp + 16)), acc);
  }
#endif
#pragma unroll
  for (int v = 0; v < 8; ++v) {
    int m = m0 + wave * 16 + v + half * 8;
    S[(size_t)m * CLS + l] = acc[v] * dv[m];
  }
}

// ---------------------------------------------------------------------------
// K6: per-row log_softmax over 16 classes (one thread per row).
__global__ __launch_bounds__(256) void k_lsm(const float* __restrict__ S,
                                             float* __restrict__ out) {
  const int row = blockIdx.x * blockDim.x + threadIdx.x;
  const float* p = S + (size_t)row * CLS;
  float x[16];
#pragma unroll
  for (int i = 0; i < 4; ++i) {
    f32x4 v = *(const f32x4*)(p + i * 4);
    x[i * 4 + 0] = v.x; x[i * 4 + 1] = v.y; x[i * 4 + 2] = v.z; x[i * 4 + 3] = v.w;
  }
  float mx = x[0];
#pragma unroll
  for (int i = 1; i < 16; ++i) mx = fmaxf(mx, x[i]);
  float sum = 0.f;
#pragma unroll
  for (int i = 0; i < 16; ++i) sum += __expf(x[i] - mx);
  float lse = mx + __logf(sum);
  float* o = out + (size_t)row * CLS;
#pragma unroll
  for (int i = 0; i < 4; ++i) {
    f32x4 v;
    v.x = x[i * 4 + 0] - lse; v.y = x[i * 4 + 1] - lse;
    v.z = x[i * 4 + 2] - lse; v.w = x[i * 4 + 3] - lse;
    *(f32x4*)(o + i * 4) = v;
  }
}

// ---------------------------------------------------------------------------
extern "C" void kernel_launch(void* const* d_in, const int* in_sizes, int n_in,
                              void* d_out, int out_size, void* d_ws, size_t ws_size,
                              hipStream_t stream) {
  const float* X  = (const float*)d_in[0];   // [8192, 512]
  const float* A  = (const float*)d_in[1];   // [8192, 8192]
  const float* W1 = (const float*)d_in[2];   // [256, 512]
  const float* W2 = (const float*)d_in[3];   // [16, 256]
  float* out = (float*)d_out;                // [8192, 16]

  char* ws = (char*)d_ws;
  size_t off = 0;
  u16*   Abf = (u16*)(ws + off);   off += (size_t)NN * NN * sizeof(u16);   // 128 MB
  float* dv  = (float*)(ws + off); off += (size_t)NN * sizeof(float);
  u16*   T1T = (u16*)(ws + off);   off += (size_t)HID * NN * sizeof(u16);  // 4 MB
  u16*   H1  = (u16*)(ws + off);   off += (size_t)NN * HID * sizeof(u16);  // 4 MB
  u16*   T2T = (u16*)(ws + off);   off += (size_t)CLS * NN * sizeof(u16);  // 256 KB
  float* S   = (float*)(ws + off); off += (size_t)NN * CLS * sizeof(float);// 512 KB
  (void)ws_size; (void)in_sizes; (void)n_in; (void)out_size;

  k_deg_cvt<<<NN, 256, 0, stream>>>(A, Abf, dv);
  k_xw1<<<(NN / 16) * (HID / 16) / 4, 128, 0, stream>>>(X, W1, dv, T1T);
  k_agg1<<<NN / 64, 256, 0, stream>>>(Abf, T1T, dv, H1);
  k_hw2<<<(NN / 16) / 4, 128, 0, stream>>>(H1, W2, dv, T2T);
  k_agg2<<<NN / 128, 256, 0, stream>>>(Abf, T2T, dv, S);
  k_lsm<<<NN / 256, 256, 0, stream>>>(S, out);
}
